// BitMGQA_19696720019655
// MI455X (gfx1250) — compile-verified, matrix operand order checked by reading, not run on hardware
//
#include <hip/hip_runtime.h>
#include <hip/hip_bf16.h>

// ---------------------------------------------------------------------------
// BitMGQA fused pipeline for gfx1250 (MI455X, wave32, WMMA)
//   B=2, T=2048, C=2048, H=16, KV=4, HD=128, KVC=512
// Pipeline:
//   f32->f16 casts -> WMMA GEMMs (Q/K/V proj, 32x64 wave tiles) ->
//   V transpose -> flash-attention (WMMA QK^T + online softmax + WMMA PV) ->
//   LayerNorm -> WMMA GEMM (output proj, f32 out)
// ---------------------------------------------------------------------------

typedef __attribute__((ext_vector_type(16))) _Float16 v16h;
typedef __attribute__((ext_vector_type(8)))  _Float16 v8h;
typedef __attribute__((ext_vector_type(8)))  float    v8f;

constexpr int Bv   = 2;
constexpr int Tv   = 2048;
constexpr int Cv   = 2048;
constexpr int Hv   = 16;
constexpr int KVv  = 4;
constexpr int HDv  = 128;   // C / H
constexpr int KVCv = 512;   // HD * KV

#define DI __device__ __forceinline__

DI v8f wmma_f16(v16h a, v16h b, v8f c) {
  // D = A(16x32 f16) x B(32x16 f16) + C(16x16 f32)
  return __builtin_amdgcn_wmma_f32_16x16x32_f16(
      /*neg_a=*/false, a, /*neg_b=*/false, b,
      /*c_mod=*/(short)0, c, /*reuse_a=*/false, /*reuse_b=*/false);
}

// Load a 16x32 f16 A-style fragment from row-major memory with leading dim ld.
// Lane l (l<16) holds row l, K = 0..7 and 16..23; lane l>=16 holds row l-16,
// K = 8..15 and 24..31 (per CDNA5 ISA 16-bit A-matrix layout). B fragments
// for X@W^T use the identical pattern with "row" = B column (W row).
DI v16h load_frag(const _Float16* __restrict__ base, int ld) {
  const int lane = threadIdx.x & 31;
  const int row  = lane & 15;
  const int khi  = (lane >> 4) << 3;          // 0 or 8
  const _Float16* p = base + (size_t)row * ld + khi;
  v8h lo = *(const v8h*)(p);
  v8h hi = *(const v8h*)(p + 16);
  v16h f;
#pragma unroll
  for (int i = 0; i < 8; ++i) { f[i] = lo[i]; f[8 + i] = hi[i]; }
  return f;
}

// ---------------------------------------------------------------------------
// f32 -> f16 cast
// ---------------------------------------------------------------------------
__global__ void __launch_bounds__(256)
cast_f32_f16(const float* __restrict__ src, _Float16* __restrict__ dst, size_t n) {
  size_t i = (size_t)blockIdx.x * blockDim.x + threadIdx.x;
  size_t stride = (size_t)gridDim.x * blockDim.x;
  for (; i < n; i += stride) dst[i] = (_Float16)src[i];
}

// ---------------------------------------------------------------------------
// GEMM: Out[M,N] = X[M,Kd] @ W[N,Kd]^T + bias[N]
// Block = 128 threads = 4 waves; block tile 128(M) x 64(N); wave tile 32x64
// (2 A fragments x 4 B fragments => each B fragment feeds 2 WMMAs).
// ---------------------------------------------------------------------------
template <bool OUT_F16>
__global__ void __launch_bounds__(128)
gemm_xwt(const _Float16* __restrict__ X, const _Float16* __restrict__ W,
         const float* __restrict__ bias, void* __restrict__ Out,
         int M, int N, int Kd) {
  const int wave = threadIdx.x >> 5;
  const int lane = threadIdx.x & 31;
  const int m0 = blockIdx.x * 128 + wave * 32;
  const int n0 = blockIdx.y * 64;

  v8f acc[2][4];
#pragma unroll
  for (int i = 0; i < 2; ++i)
#pragma unroll
    for (int s = 0; s < 4; ++s) {
      v8f z = {0.f, 0.f, 0.f, 0.f, 0.f, 0.f, 0.f, 0.f};
      acc[i][s] = z;
    }

  const int frow = lane & 15;
  for (int kc = 0; kc < Kd; kc += 32) {
    if (kc + 64 <= Kd) {  // prefetch next K-chunk (global_prefetch_b8)
      __builtin_prefetch(X + (size_t)(m0 + frow) * Kd + kc + 32, 0, 3);
      __builtin_prefetch(W + (size_t)(n0 + frow) * Kd + kc + 32, 0, 3);
    }
    v16h a0 = load_frag(X + (size_t)m0 * Kd + kc, Kd);
    v16h a1 = load_frag(X + (size_t)(m0 + 16) * Kd + kc, Kd);
#pragma unroll
    for (int s = 0; s < 4; ++s) {
      v16h b = load_frag(W + (size_t)(n0 + s * 16) * Kd + kc, Kd);
      acc[0][s] = wmma_f16(a0, b, acc[0][s]);
      acc[1][s] = wmma_f16(a1, b, acc[1][s]);
    }
  }

  // Epilogue: C layout — VGPR r holds M = m0+r (lanes 0-15) / m0+r+8 (16-31),
  // N = n0 + sub*16 + lane%16.
  const int col   = lane & 15;
  const int rbase = (lane >> 4) << 3;
#pragma unroll
  for (int i = 0; i < 2; ++i) {
#pragma unroll
    for (int s = 0; s < 4; ++s) {
      const float bval = bias[n0 + s * 16 + col];
#pragma unroll
      for (int r = 0; r < 8; ++r) {
        const size_t idx =
            (size_t)(m0 + i * 16 + r + rbase) * N + n0 + s * 16 + col;
        const float v = acc[i][s][r] + bval;
        if constexpr (OUT_F16) ((_Float16*)Out)[idx] = (_Float16)v;
        else                   ((float*)Out)[idx] = v;
      }
    }
  }
}

// ---------------------------------------------------------------------------
// V transpose: vp [B,T,KVC] -> vt [B,KV,HD,T]  (makes PV B-fragments K-contiguous)
// ---------------------------------------------------------------------------
__global__ void __launch_bounds__(256)
transpose_v(const _Float16* __restrict__ vp, _Float16* __restrict__ vt) {
  const size_t n = (size_t)Bv * Tv * KVCv;
  size_t i = (size_t)blockIdx.x * blockDim.x + threadIdx.x;
  if (i >= n) return;
  const int c = (int)(i % KVCv);
  const size_t bt = i / KVCv;
  const int t = (int)(bt % Tv);
  const int b = (int)(bt / Tv);
  const int g = c / HDv;
  const int d = c % HDv;
  vt[(((size_t)(b * KVv + g)) * HDv + d) * Tv + t] = vp[i];
}

// ---------------------------------------------------------------------------
// Flash attention: one wave per 16 query rows of one (b,h).
//   Q  : [B,T,C]   f16 (head h slice at h*HD)
//   Kp : [B,T,KVC] f16 (group g slice at g*HD)
//   Vt : [B,KV,HD,T] f16
//   Y  : [B,T,C]   f32
// ---------------------------------------------------------------------------
__global__ void __launch_bounds__(128)
attn_flash(const _Float16* __restrict__ Q, const _Float16* __restrict__ Kp,
           const _Float16* __restrict__ Vt, float* __restrict__ Y) {
  const int wave = threadIdx.x >> 5;
  const int lane = threadIdx.x & 31;
  const int bh = blockIdx.y;             // 0 .. B*H-1
  const int b = bh / Hv;
  const int h = bh % Hv;
  const int g = h / (Hv / KVv);          // GQA group (repeat_interleave)
  const int t0 = (blockIdx.x * 4 + wave) * 16;

  __shared__ __align__(16) _Float16 plds_all[4][16 * 32];
  _Float16* plds = plds_all[wave];

  // Q fragments: 16 rows x 128, as 4 chunks of K=32 (resident in VGPRs)
  const _Float16* qbase = Q + ((size_t)(b * Tv + t0)) * Cv + h * HDv;
  v16h qf[4];
#pragma unroll
  for (int c = 0; c < 4; ++c) qf[c] = load_frag(qbase + c * 32, Cv);

  v8f y[8];
#pragma unroll
  for (int d = 0; d < 8; ++d) {
    v8f z = {0.f, 0.f, 0.f, 0.f, 0.f, 0.f, 0.f, 0.f};
    y[d] = z;
  }
  float m_run[8], l_run[8];
#pragma unroll
  for (int r = 0; r < 8; ++r) { m_run[r] = -1e30f; l_run[r] = 0.0f; }

  const float scale = 0.088388347648318447f;   // 1/sqrt(128)
  const int frow = lane & 15;

  for (int kt = 0; kt < Tv; kt += 32) {
    const _Float16* kbase = Kp + ((size_t)(b * Tv + kt)) * KVCv + g * HDv;
    const _Float16* vbase = Vt + (((size_t)(b * KVv + g)) * HDv) * Tv + kt;
    if (kt + 64 <= Tv) {  // prefetch next K/V tiles
      __builtin_prefetch(kbase + (size_t)32 * KVCv + (size_t)frow * KVCv, 0, 3);
      __builtin_prefetch(vbase + 32 + (size_t)frow * Tv, 0, 3);
    }

    // ---- S = Q @ K^T : 16 x 32 (two 16-key column groups) ----
    v8f s0 = {0.f, 0.f, 0.f, 0.f, 0.f, 0.f, 0.f, 0.f};
    v8f s1 = {0.f, 0.f, 0.f, 0.f, 0.f, 0.f, 0.f, 0.f};
#pragma unroll
    for (int c = 0; c < 4; ++c) {
      v16h k0 = load_frag(kbase + c * 32, KVCv);                     // keys kt..kt+15
      v16h k1 = load_frag(kbase + (size_t)16 * KVCv + c * 32, KVCv); // keys kt+16..kt+31
      s0 = wmma_f16(qf[c], k0, s0);
      s1 = wmma_f16(qf[c], k1, s1);
    }

    // ---- online softmax over the 32-key tile ----
    // C-layout: VGPR r -> row r (lanes 0-15) / row r+8 (lanes 16-31);
    // columns live across the 16 lanes of each half-wave, so xor-shuffles
    // with masks 1..8 reduce exactly one row.
#pragma unroll
    for (int r = 0; r < 8; ++r) {
      const float a0 = s0[r] * scale;
      const float a1 = s1[r] * scale;
      float mx = fmaxf(a0, a1);
#pragma unroll
      for (int off = 8; off >= 1; off >>= 1) mx = fmaxf(mx, __shfl_xor(mx, off, 32));
      const float mnew  = fmaxf(m_run[r], mx);
      const float alpha = __expf(m_run[r] - mnew);
      const float p0 = __expf(a0 - mnew);
      const float p1 = __expf(a1 - mnew);
      float sum = p0 + p1;
#pragma unroll
      for (int off = 8; off >= 1; off >>= 1) sum += __shfl_xor(sum, off, 32);
      l_run[r] = l_run[r] * alpha + sum;
      m_run[r] = mnew;
#pragma unroll
      for (int d = 0; d < 8; ++d) y[d][r] *= alpha;
      // Stash P (f16) in LDS to convert C-layout -> A-layout
      const int row = r + ((lane >> 4) << 3);
      const int col = lane & 15;
      plds[row * 32 + col]      = (_Float16)p0;
      plds[row * 32 + col + 16] = (_Float16)p1;
    }
    asm volatile("s_wait_dscnt 0" ::: "memory");

    // ---- y += P @ V_tile ----
    v16h pf = load_frag(plds, 32);  // P as 16x32 A fragment
#pragma unroll
    for (int d = 0; d < 8; ++d) {
      v16h vf = load_frag(vbase + (size_t)(d * 16) * Tv, Tv);  // B col n = hd d*16+n
      y[d] = wmma_f16(pf, vf, y[d]);
    }
  }

  // ---- finalize: divide by l, scatter to [B,T,C] f32 ----
  const int col   = lane & 15;
  const int rbase = (lane >> 4) << 3;
#pragma unroll
  for (int r = 0; r < 8; ++r) {
    const float inv = 1.0f / l_run[r];
    const size_t rowoff = ((size_t)(b * Tv + t0 + r + rbase)) * Cv + h * HDv + col;
#pragma unroll
    for (int d = 0; d < 8; ++d) Y[rowoff + d * 16] = y[d][r] * inv;
  }
}

// ---------------------------------------------------------------------------
// LayerNorm over C, f32 in -> f16 out (scaled/shifted)
// ---------------------------------------------------------------------------
__global__ void __launch_bounds__(256)
layernorm(const float* __restrict__ Yin, const float* __restrict__ lnw,
          const float* __restrict__ lnb, _Float16* __restrict__ Xout) {
  const int row = blockIdx.x;
  const float* x = Yin + (size_t)row * Cv;
  float s = 0.f, s2 = 0.f;
  for (int i = threadIdx.x; i < Cv; i += 256) {
    const float v = x[i];
    s += v; s2 += v * v;
  }
#pragma unroll
  for (int off = 16; off >= 1; off >>= 1) {
    s  += __shfl_xor(s, off, 32);
    s2 += __shfl_xor(s2, off, 32);
  }
  __shared__ float ws[8], ws2[8];
  if ((threadIdx.x & 31) == 0) {
    ws[threadIdx.x >> 5] = s;
    ws2[threadIdx.x >> 5] = s2;
  }
  __syncthreads();
  float ts = 0.f, ts2 = 0.f;
#pragma unroll
  for (int i = 0; i < 8; ++i) { ts += ws[i]; ts2 += ws2[i]; }
  const float mu   = ts / Cv;
  const float var  = ts2 / Cv - mu * mu;
  const float rstd = rsqrtf(var + 1e-5f);
  _Float16* o = Xout + (size_t)row * Cv;
  for (int i = threadIdx.x; i < Cv; i += 256)
    o[i] = (_Float16)((x[i] - mu) * rstd * lnw[i] + lnb[i]);
}

// ---------------------------------------------------------------------------
// Host orchestration
// ---------------------------------------------------------------------------
extern "C" void kernel_launch(void* const* d_in, const int* in_sizes, int n_in,
                              void* d_out, int out_size, void* d_ws, size_t ws_size,
                              hipStream_t stream) {
  (void)in_sizes; (void)n_in; (void)out_size; (void)ws_size;
  const float* query = (const float*)d_in[0];
  const float* key   = (const float*)d_in[1];
  const float* value = (const float*)d_in[2];
  const float* Wq    = (const float*)d_in[3];
  const float* bq    = (const float*)d_in[4];
  const float* Wk    = (const float*)d_in[5];
  const float* bk    = (const float*)d_in[6];
  const float* Wv    = (const float*)d_in[7];
  const float* bv    = (const float*)d_in[8];
  const float* ln_w  = (const float*)d_in[9];
  const float* ln_b  = (const float*)d_in[10];
  const float* Wo    = (const float*)d_in[11];
  const float* bo    = (const float*)d_in[12];
  float* out = (float*)d_out;

  const size_t nBTC  = (size_t)Bv * Tv * Cv;      // 8,388,608
  const size_t nBTKV = (size_t)Bv * Tv * KVCv;    // 2,097,152
  const size_t nCC   = (size_t)Cv * Cv;           // 4,194,304
  const size_t nKC   = (size_t)KVCv * Cv;         // 1,048,576

  char* w = (char*)d_ws;
  auto take = [&](size_t bytes) {
    char* p = w;
    w += (bytes + 255) & ~(size_t)255;
    return p;
  };
  _Float16* q16  = (_Float16*)take(nBTC * 2);
  _Float16* k16  = (_Float16*)take(nBTC * 2);
  _Float16* v16  = (_Float16*)take(nBTC * 2);
  _Float16* wq16 = (_Float16*)take(nCC * 2);
  _Float16* wk16 = (_Float16*)take(nKC * 2);
  _Float16* wv16 = (_Float16*)take(nKC * 2);
  _Float16* wo16 = (_Float16*)take(nCC * 2);
  _Float16* qp   = (_Float16*)take(nBTC * 2);
  _Float16* kp   = (_Float16*)take(nBTKV * 2);
  _Float16* vp   = (_Float16*)take(nBTKV * 2);
  _Float16* vt   = (_Float16*)take(nBTKV * 2);
  float*    yat  = (float*)take(nBTC * 4);
  _Float16* xln  = (_Float16*)take(nBTC * 2);

  auto cast = [&](const float* s, _Float16* d, size_t n) {
    int blocks = (int)((n + 255) / 256);
    if (blocks > 16384) blocks = 16384;
    cast_f32_f16<<<blocks, 256, 0, stream>>>(s, d, n);
  };
  cast(query, q16, nBTC);
  cast(key,   k16, nBTC);
  cast(value, v16, nBTC);
  cast(Wq, wq16, nCC);
  cast(Wk, wk16, nKC);
  cast(Wv, wv16, nKC);
  cast(Wo, wo16, nCC);

  const int M = Bv * Tv;  // 4096

  // QKV projections (block tile 128x64)
  gemm_xwt<true><<<dim3(M / 128, Cv / 64), 128, 0, stream>>>(
      q16, wq16, bq, qp, M, Cv, Cv);
  gemm_xwt<true><<<dim3(M / 128, KVCv / 64), 128, 0, stream>>>(
      k16, wk16, bk, kp, M, KVCv, Cv);
  gemm_xwt<true><<<dim3(M / 128, KVCv / 64), 128, 0, stream>>>(
      v16, wv16, bv, vp, M, KVCv, Cv);

  // V transpose for PV fragment loads
  transpose_v<<<(int)((nBTKV + 255) / 256), 256, 0, stream>>>(vp, vt);

  // Flash attention
  attn_flash<<<dim3(Tv / 64, Bv * Hv), 128, 0, stream>>>(qp, kp, vt, yat);

  // LayerNorm
  layernorm<<<M, 256, 0, stream>>>(yat, ln_w, ln_b, xln);

  // Output projection (f32 out)
  gemm_xwt<false><<<dim3(M / 128, Cv / 64), 128, 0, stream>>>(
      xln, wo16, bo, out, M, Cv, Cv);
}
